// DreamFitSingleStreamProcessor_9448928051278
// MI455X (gfx1250) — matrix operand order repaired and checked by reading, not compiled, not run on hardware
//
#include <hip/hip_runtime.h>
#include <hip/hip_bf16.h>
#include <cstddef>

typedef __attribute__((ext_vector_type(16))) __bf16 v16bf;
typedef __attribute__((ext_vector_type(8)))  float  v8f;
typedef __attribute__((ext_vector_type(4)))  float  v4f;
typedef __attribute__((ext_vector_type(4)))  __bf16 v4bf;
typedef __attribute__((ext_vector_type(2)))  float  v2f;
typedef __attribute__((ext_vector_type(2)))  __bf16 v2bf;
typedef unsigned short u16;
typedef unsigned int   u32;

#define HID   3072
#define HEADS 24
#define HD    128
#define MLP   12288
#define SEQ   2048
#define W1OUT (3*HID + MLP)      // 21504
#define CATW  (HID + MLP)        // 15360

union FragBF { v16bf v; u32 u[8]; uint4 q[2]; };

// hardware packed float->bf16 (RNE); lowers to v_cvt_pk_bf16_f32, no repack
__device__ __forceinline__ u16 f2bf(float f) {
  return __builtin_bit_cast(u16, (__bf16)f);
}
__device__ __forceinline__ u32 pack2(float a, float b) {
  v2f t = {a, b};
  return __builtin_bit_cast(u32, __builtin_convertvector(t, v2bf));
}
__device__ __forceinline__ uint2 pack4(float4 v) {
  v4f t = {v.x, v.y, v.z, v.w};
  return __builtin_bit_cast(uint2, __builtin_convertvector(t, v4bf));
}
__device__ __forceinline__ v8f wmma_bf16(const FragBF& a, const FragBF& b, v8f c) {
  return __builtin_amdgcn_wmma_f32_16x16x32_bf16(false, a.v, false, b.v,
                                                 (short)0, c, false, false);
}

// ---------------------------------------------------------------------------
// m = silu(vec) @ mod_w.T + mod_b   (one wave per output)
// ---------------------------------------------------------------------------
__global__ __launch_bounds__(256)
void k_mod(const float* __restrict__ vec, const float* __restrict__ mod_w,
           const float* __restrict__ mod_b, float* __restrict__ m)
{
  int lane = threadIdx.x & 31, wid = threadIdx.x >> 5;
  int o = blockIdx.x * 8 + wid;
  float s = 0.0f;
  for (int j = lane; j < HID; j += 32) {
    float v = vec[j];
    float sv = v / (1.0f + __expf(-v));
    s += sv * mod_w[(size_t)o * HID + j];
  }
  #pragma unroll
  for (int off = 16; off > 0; off >>= 1) s += __shfl_xor(s, off, 32);
  if (lane == 0) m[o] = s + mod_b[o];
}

// ---------------------------------------------------------------------------
// x_mod = (1+scale)*layernorm(x) + shift   (one block per row)
// ---------------------------------------------------------------------------
__global__ __launch_bounds__(256)
void k_lnmod(const float* __restrict__ x, const float* __restrict__ m,
             float* __restrict__ xmod)
{
  __shared__ float rs_[256], rq_[256];
  int row = blockIdx.x, tid = threadIdx.x;
  const float* xr = x + (size_t)row * HID;
  float s = 0.0f, q = 0.0f;
  for (int c = tid; c < HID; c += 256) { float v = xr[c]; s += v; q += v * v; }
  rs_[tid] = s; rq_[tid] = q; __syncthreads();
  for (int st = 128; st > 0; st >>= 1) {
    if (tid < st) { rs_[tid] += rs_[tid + st]; rq_[tid] += rq_[tid + st]; }
    __syncthreads();
  }
  float mu  = rs_[0] * (1.0f / HID);
  float var = rq_[0] * (1.0f / HID) - mu * mu;
  float inv = rsqrtf(var + 1e-6f);
  for (int c = tid; c < HID; c += 256) {
    float shift = m[c], scale = m[HID + c];
    xmod[(size_t)row * HID + c] = (1.0f + scale) * ((xr[c] - mu) * inv) + shift;
  }
}

// ---------------------------------------------------------------------------
// Generic mixed-precision GEMM: C(MxN) = A(MxK,f32) @ B(NxK,f32)^T
// fp32->bf16 via packed hardware cvt while staging to LDS; wmma bf16, f32 acc.
// 128x128x32 block tile; 256 threads = 8 waves, each wave 2x4 16x16 tiles.
// Grid: blockIdx.x = M-tile (fast -> B panels reused through L2), .y = N-tile.
// Double-buffered LDS, register staging, global prefetch 2 stages ahead.
// accumulate=0: C = acc + bias[col] (bias may be null) ; =1: C += acc
// ---------------------------------------------------------------------------
__global__ __launch_bounds__(256)
void k_gemm(const float* __restrict__ A, int lda,
            const float* __restrict__ B, int ldb,
            float* __restrict__ C, int ldc,
            const float* __restrict__ bias,
            int M, int N, int K, int accumulate)
{
  __shared__ __align__(16) u16 sA[2][128][40];
  __shared__ __align__(16) u16 sB[2][128][40];

  const int tid  = threadIdx.x;
  const int lane = tid & 31, wid = tid >> 5;
  const int wm   = wid & 3,  wn  = wid >> 2;
  const int half = lane >> 4, lm = lane & 15;
  const int m0 = blockIdx.x * 128;   // M fast: consecutive blocks share B panel
  const int n0 = blockIdx.y * 128;

  const int r_ld  = tid >> 3;          // 0..31 (row within 32-row chunk)
  const int kk_ld = (tid & 7) << 2;    // 0,4,..,28

  v8f acc[2][4];
  #pragma unroll
  for (int i = 0; i < 2; i++)
    #pragma unroll
    for (int j = 0; j < 4; j++) acc[i][j] = (v8f)(0.0f);

  float4 pa[4], pb[4];
  const float4 z4 = make_float4(0.f, 0.f, 0.f, 0.f);

  // ---- stage 0 loads ----
  #pragma unroll
  for (int t = 0; t < 4; t++) {
    int r = t * 32 + r_ld;
    pa[t] = *(const float4*)(A + (size_t)(m0 + r) * lda + kk_ld);
    pb[t] = (n0 + r < N) ? *(const float4*)(B + (size_t)(n0 + r) * ldb + kk_ld) : z4;
  }
  #pragma unroll
  for (int t = 0; t < 4; t++) {
    int r = t * 32 + r_ld;
    *(uint2*)&sA[0][r][kk_ld] = pack4(pa[t]);
    *(uint2*)&sB[0][r][kk_ld] = pack4(pb[t]);
  }
  __syncthreads();

  int stage = 0;
  for (int k0 = 0; k0 < K; k0 += 32, stage ^= 1) {
    const bool has_next = (k0 + 32 < K);

    // prefetch 2 stages ahead into L2 (global_prefetch_b8)
    if (k0 + 64 < K) {
      __builtin_prefetch(A + (size_t)(m0 + r_ld) * lda + k0 + 64 + kk_ld, 0, 1);
      if (n0 + r_ld < N)
        __builtin_prefetch(B + (size_t)(n0 + r_ld) * ldb + k0 + 64 + kk_ld, 0, 1);
    }

    // issue next stage global loads into registers
    if (has_next) {
      #pragma unroll
      for (int t = 0; t < 4; t++) {
        int r = t * 32 + r_ld;
        pa[t] = *(const float4*)(A + (size_t)(m0 + r) * lda + (k0 + 32) + kk_ld);
        pb[t] = (n0 + r < N) ? *(const float4*)(B + (size_t)(n0 + r) * ldb + (k0 + 32) + kk_ld) : z4;
      }
    }

    // compute current stage from LDS
    {
      FragBF a[2], b[4];
      #pragma unroll
      for (int i = 0; i < 2; i++) {          // A 16x32: lanes 0-15 K0-15, 16-31 K16-31
        const u16* p = &sA[stage][wm * 32 + i * 16 + lm][half * 16];
        a[i].q[0] = *(const uint4*)p;
        a[i].q[1] = *(const uint4*)(p + 8);
      }
      #pragma unroll
      for (int j = 0; j < 4; j++) {          // B 32x16: lanes 0-15 K0-15, 16-31 K16-31
        const u16* p = &sB[stage][wn * 64 + j * 16 + lm][half * 16];
        b[j].q[0] = *(const uint4*)p;
        b[j].q[1] = *(const uint4*)(p + 8);
      }
      #pragma unroll
      for (int i = 0; i < 2; i++)
        #pragma unroll
        for (int j = 0; j < 4; j++)
          acc[i][j] = wmma_bf16(a[i], b[j], acc[i][j]);
    }

    // store next stage into the other LDS buffer (not read this iteration)
    if (has_next) {
      int ns = stage ^ 1;
      #pragma unroll
      for (int t = 0; t < 4; t++) {
        int r = t * 32 + r_ld;
        *(uint2*)&sA[ns][r][kk_ld] = pack4(pa[t]);
        *(uint2*)&sB[ns][r][kk_ld] = pack4(pb[t]);
      }
    }
    __syncthreads();
  }

  #pragma unroll
  for (int i = 0; i < 2; i++) {
    #pragma unroll
    for (int j = 0; j < 4; j++) {
      int col = n0 + wn * 64 + j * 16 + lm;
      if (col < N) {
        int rowb = m0 + wm * 32 + i * 16 + half * 8;
        float bv = bias ? bias[col] : 0.0f;
        #pragma unroll
        for (int r = 0; r < 8; r++) {
          size_t idx = (size_t)(rowb + r) * ldc + col;
          float v = acc[i][j][r];
          if (accumulate) v += C[idx]; else v += bv;
          C[idx] = v;
        }
      }
    }
  }
}

// ---------------------------------------------------------------------------
// In-place RMSnorm + RoPE on q,k inside h; q prescaled by 1/sqrt(HD).
// grid (L, HEADS), 64 threads (one per d-pair).
// ---------------------------------------------------------------------------
__global__ __launch_bounds__(64)
void k_qkv_prep(float* __restrict__ h, const float* __restrict__ pe,
                const float* __restrict__ q_scale, const float* __restrict__ k_scale)
{
  __shared__ float red[64];
  int l = blockIdx.x, head = blockIdx.y, e = threadIdx.x;
  float* base = h + (size_t)l * W1OUT + head * HD;
  float qe = base[2 * e],       qo = base[2 * e + 1];
  float ke = base[HID + 2 * e], ko = base[HID + 2 * e + 1];

  red[e] = qe * qe + qo * qo; __syncthreads();
  for (int s = 32; s > 0; s >>= 1) { if (e < s) red[e] += red[e + s]; __syncthreads(); }
  float qr = rsqrtf(red[0] * (1.0f / HD) + 1e-6f);
  __syncthreads();
  red[e] = ke * ke + ko * ko; __syncthreads();
  for (int s = 32; s > 0; s >>= 1) { if (e < s) red[e] += red[e + s]; __syncthreads(); }
  float kr = rsqrtf(red[0] * (1.0f / HD) + 1e-6f);

  qe *= qr * q_scale[2 * e]; qo *= qr * q_scale[2 * e + 1];
  ke *= kr * k_scale[2 * e]; ko *= kr * k_scale[2 * e + 1];

  const float* p = pe + ((size_t)l * (HD / 2) + e) * 4;   // [cos,-sin,sin,cos]
  const float sc = 0.08838834764831845f;                  // 1/sqrt(128)
  base[2 * e]           = (p[0] * qe + p[1] * qo) * sc;
  base[2 * e + 1]       = (p[2] * qe + p[3] * qo) * sc;
  base[HID + 2 * e]     =  p[0] * ke + p[1] * ko;
  base[HID + 2 * e + 1] =  p[2] * ke + p[3] * ko;
}

// ---------------------------------------------------------------------------
// Flash attention: one block = (head, 64 q-rows). 256 threads = 8 waves.
// Q frags persist in registers; K/V^T tiles bf16 in LDS; S & P via WMMA.
// Writes attn_out into cat[:, head*HD ...] (row stride CATW).
// ---------------------------------------------------------------------------
__global__ __launch_bounds__(256)
void k_attention(const float* __restrict__ h, float* __restrict__ cat)
{
  __shared__ __align__(16) u16   sK[64][136];
  __shared__ __align__(16) u16   sVt[128][72];
  __shared__ __align__(16) float sS[64][68];
  __shared__ __align__(16) u16   sP[64][72];
  __shared__ float sAlpha[64];
  __shared__ float sL[64];

  const int tid  = threadIdx.x;
  const int lane = tid & 31, wid = tid >> 5;
  const int half = lane >> 4, lm = lane & 15;
  const int tm = wid >> 1;          // row tile 0..3
  const int wn = wid & 1;           // col half
  const int head = blockIdx.y;
  const int qb0  = blockIdx.x * 64;

  const float* Hq = h + head * HD;
  const float* Hk = h + HID + head * HD;
  const float* Hv = h + 2 * HID + head * HD;

  // persistent Q fragments: 4 x (16x32) over d
  FragBF qf[4];
  {
    const float* qrow = Hq + (size_t)(qb0 + tm * 16 + lm) * W1OUT;
    #pragma unroll
    for (int f = 0; f < 4; f++) {
      #pragma unroll
      for (int j = 0; j < 4; j++) {
        int d0 = f * 32 + half * 16 + 2 * j;
        qf[f].u[j]     = pack2(qrow[d0], qrow[d0 + 1]);
        int d1 = f * 32 + half * 16 + 8 + 2 * j;
        qf[f].u[4 + j] = pack2(qrow[d1], qrow[d1 + 1]);
      }
    }
  }

  v8f oc[4];
  #pragma unroll
  for (int t = 0; t < 4; t++) oc[t] = (v8f)(0.0f);
  float m_i = -1e30f, l_i = 0.0f;

  for (int kb0 = 0; kb0 < SEQ; kb0 += 64) {
    // stage K tile (rows=key, contiguous d) and V^T tile
    #pragma unroll
    for (int t = 0; t < 8; t++) {
      int idx = t * 256 + tid;       // 0..2047
      int r  = idx >> 5;             // key row 0..63
      int kk = (idx & 31) << 2;      // d 0..124
      float4 kv = *(const float4*)(Hk + (size_t)(kb0 + r) * W1OUT + kk);
      *(uint2*)&sK[r][kk] = pack4(kv);
      float4 vv = *(const float4*)(Hv + (size_t)(kb0 + r) * W1OUT + kk);
      sVt[kk + 0][r] = f2bf(vv.x);
      sVt[kk + 1][r] = f2bf(vv.y);
      sVt[kk + 2][r] = f2bf(vv.z);
      sVt[kk + 3][r] = f2bf(vv.w);
    }
    __syncthreads();

    // S = Q K^T : each wave computes tiles (tm, 2*wn) and (tm, 2*wn+1)
    {
      v8f s0 = (v8f)(0.0f), s1 = (v8f)(0.0f);
      #pragma unroll
      for (int ks = 0; ks < 4; ks++) {
        FragBF b0, b1;
        const u16* p0 = &sK[(2 * wn + 0) * 16 + lm][ks * 32 + half * 16];
        b0.q[0] = *(const uint4*)p0; b0.q[1] = *(const uint4*)(p0 + 8);
        const u16* p1 = &sK[(2 * wn + 1) * 16 + lm][ks * 32 + half * 16];
        b1.q[0] = *(const uint4*)p1; b1.q[1] = *(const uint4*)(p1 + 8);
        s0 = wmma_bf16(qf[ks], b0, s0);
        s1 = wmma_bf16(qf[ks], b1, s1);
      }
      #pragma unroll
      for (int r = 0; r < 8; r++) {
        int rowl = tm * 16 + half * 8 + r;
        sS[rowl][(2 * wn + 0) * 16 + lm] = s0[r];
        sS[rowl][(2 * wn + 1) * 16 + lm] = s1[r];
      }
    }
    __syncthreads();

    // online softmax: one thread per row
    if (tid < 64) {
      float mrow = -1e30f;
      for (int j = 0; j < 64; j++) mrow = fmaxf(mrow, sS[tid][j]);
      float mnew  = fmaxf(m_i, mrow);
      float alpha = __expf(m_i - mnew);
      float s = 0.0f;
      for (int j = 0; j < 64; j++) {
        float pj = __expf(sS[tid][j] - mnew);
        s += pj;
        sP[tid][j] = f2bf(pj);
      }
      l_i = l_i * alpha + s;
      m_i = mnew;
      sAlpha[tid] = alpha;
    }
    __syncthreads();

    // O = alpha*O + P @ V : wave covers rows tm*16.., cols wn*64..+64
    {
      float af[8];
      #pragma unroll
      for (int r = 0; r < 8; r++) af[r] = sAlpha[tm * 16 + half * 8 + r];
      #pragma unroll
      for (int t = 0; t < 4; t++)
        #pragma unroll
        for (int r = 0; r < 8; r++) oc[t][r] *= af[r];

      FragBF pa[2];
      #pragma unroll
      for (int kk = 0; kk < 2; kk++) {
        const u16* p = &sP[tm * 16 + lm][kk * 32 + half * 16];
        pa[kk].q[0] = *(const uint4*)p;
        pa[kk].q[1] = *(const uint4*)(p + 8);
      }
      #pragma unroll
      for (int t = 0; t < 4; t++) {
        #pragma unroll
        for (int kk = 0; kk < 2; kk++) {
          FragBF vb;
          const u16* p = &sVt[wn * 64 + t * 16 + lm][kk * 32 + half * 16];
          vb.q[0] = *(const uint4*)p;
          vb.q[1] = *(const uint4*)(p + 8);
          oc[t] = wmma_bf16(pa[kk], vb, oc[t]);
        }
      }
    }
    __syncthreads();
  }

  if (tid < 64) sL[tid] = l_i;
  __syncthreads();

  #pragma unroll
  for (int r = 0; r < 8; r++) {
    int rowl = tm * 16 + half * 8 + r;
    float inv = 1.0f / sL[rowl];
    size_t grow = (size_t)(qb0 + rowl) * CATW;
    #pragma unroll
    for (int t = 0; t < 4; t++) {
      int col = head * HD + wn * 64 + t * 16 + lm;
      cat[grow + col] = oc[t][r] * inv;
    }
  }
}

// ---------------------------------------------------------------------------
__global__ __launch_bounds__(256)
void k_gelu_cat(const float* __restrict__ h, float* __restrict__ cat)
{
  size_t idx = (size_t)blockIdx.x * 256 + threadIdx.x;
  size_t l = idx / MLP, c = idx % MLP;
  float v = h[l * W1OUT + 3 * HID + c];
  float g = 0.5f * v * (1.0f + tanhf(0.7978845608028654f * (v + 0.044715f * v * v * v)));
  cat[l * CATW + HID + c] = g;
}

__global__ __launch_bounds__(256)
void k_residual(const float* __restrict__ x, const float* __restrict__ m,
                const float* __restrict__ outb, float* __restrict__ out)
{
  size_t idx = (size_t)blockIdx.x * 256 + threadIdx.x;
  size_t c = idx % HID;
  out[idx] = x[idx] + m[2 * HID + c] * outb[idx];
}

// ---------------------------------------------------------------------------
extern "C" void kernel_launch(void* const* d_in, const int* in_sizes, int n_in,
                              void* d_out, int out_size, void* d_ws, size_t ws_size,
                              hipStream_t stream)
{
  const float* x        = (const float*)d_in[0];
  const float* vec      = (const float*)d_in[1];
  const float* pe       = (const float*)d_in[2];
  const float* mod_w    = (const float*)d_in[3];
  const float* mod_b    = (const float*)d_in[4];
  const float* w1       = (const float*)d_in[5];
  const float* b1       = (const float*)d_in[6];
  const float* w2       = (const float*)d_in[7];
  const float* b2       = (const float*)d_in[8];
  const float* q_scale  = (const float*)d_in[9];
  const float* k_scale  = (const float*)d_in[10];
  const float* lqd      = (const float*)d_in[11];
  const float* lqu      = (const float*)d_in[12];
  const float* lkd      = (const float*)d_in[13];
  const float* lku      = (const float*)d_in[14];
  const float* lvd      = (const float*)d_in[15];
  const float* lvu      = (const float*)d_in[16];
  const float* pdn      = (const float*)d_in[17];
  const float* pup      = (const float*)d_in[18];

  float* W    = (float*)d_ws;
  float* mbuf = W;                                   //   9216
  float* xmod = W + 16384;                           // 2048*3072
  float* h    = xmod + (size_t)SEQ * HID;            // 2048*21504
  float* tmp  = h + (size_t)SEQ * W1OUT;             // 2048*64
  float* cat  = tmp + (size_t)SEQ * 64;              // 2048*15360
  float* outb = cat + (size_t)SEQ * CATW;            // 2048*3072

  // 1) modulation vector
  k_mod<<<(3 * HID) / 8, 256, 0, stream>>>(vec, mod_w, mod_b, mbuf);
  // 2) layernorm + modulation
  k_lnmod<<<SEQ, 256, 0, stream>>>(x, mbuf, xmod);
  // 3) h = x_mod @ w1.T + b1   (M fast dim -> weight panels L2-resident)
  k_gemm<<<dim3(SEQ / 128, W1OUT / 128), 256, 0, stream>>>(
      xmod, HID, w1, HID, h, W1OUT, b1, SEQ, W1OUT, HID, 0);
  // 4) LoRA q/k/v (down then accumulate up into h's q/k/v columns)
  k_gemm<<<dim3(SEQ / 128, 1), 256, 0, stream>>>(xmod, HID, lqd, HID, tmp, 32, nullptr, SEQ, 32, HID, 0);
  k_gemm<<<dim3(SEQ / 128, HID / 128), 256, 0, stream>>>(tmp, 32, lqu, 32, h, W1OUT, nullptr, SEQ, HID, 32, 1);
  k_gemm<<<dim3(SEQ / 128, 1), 256, 0, stream>>>(xmod, HID, lkd, HID, tmp, 32, nullptr, SEQ, 32, HID, 0);
  k_gemm<<<dim3(SEQ / 128, HID / 128), 256, 0, stream>>>(tmp, 32, lku, 32, h + HID, W1OUT, nullptr, SEQ, HID, 32, 1);
  k_gemm<<<dim3(SEQ / 128, 1), 256, 0, stream>>>(xmod, HID, lvd, HID, tmp, 32, nullptr, SEQ, 32, HID, 0);
  k_gemm<<<dim3(SEQ / 128, HID / 128), 256, 0, stream>>>(tmp, 32, lvu, 32, h + 2 * HID, W1OUT, nullptr, SEQ, HID, 32, 1);
  // 5) rms + rope (+ q prescale) in place
  k_qkv_prep<<<dim3(SEQ, HEADS), 64, 0, stream>>>(h, pe, q_scale, k_scale);
  // 6) attention -> cat[:, :HID]
  k_attention<<<dim3(SEQ / 64, HEADS), 256, 0, stream>>>(h, cat);
  // 7) gelu(mlp) -> cat[:, HID:]
  k_gelu_cat<<<((size_t)SEQ * MLP) / 256, 256, 0, stream>>>(h, cat);
  // 8) out = cat @ w2.T + b2
  k_gemm<<<dim3(SEQ / 128, HID / 128), 256, 0, stream>>>(
      cat, CATW, w2, CATW, outb, HID, b2, SEQ, HID, CATW, 0);
  // 9) out += (cat @ proj_down.T) @ proj_up.T
  k_gemm<<<dim3(SEQ / 128, 1), 256, 0, stream>>>(cat, CATW, pdn, CATW, tmp, 64, nullptr, SEQ, 64, CATW, 0);
  k_gemm<<<dim3(SEQ / 128, HID / 128), 256, 0, stream>>>(tmp, 64, pup, 64, outb, HID, nullptr, SEQ, HID, 64, 1);
  // 10) final residual with gate
  k_residual<<<((size_t)SEQ * HID) / 256, 256, 0, stream>>>(x, mbuf, outb, (float*)d_out);
}
